// LocalAttentionBlock_137438953872
// MI455X (gfx1250) — matrix operand order repaired and checked by reading, not compiled, not run on hardware
//
#include <hip/hip_runtime.h>
#include <math.h>

#define Bsz 4
#define Tt  2048
#define Dd  768
#define Hh  12
#define HDim 64
#define MLPD 3072
#define NTOK (Bsz*Tt)   // 8192

typedef __attribute__((ext_vector_type(16))) _Float16 v16h;
typedef __attribute__((ext_vector_type(8)))  float    v8f;

union Frag16 { v16h v; uint4 q[2]; };

static __device__ __forceinline__ v16h load_frag(const _Float16* p0, const _Float16* p1) {
    Frag16 u;
    u.q[0] = *(const uint4*)p0;
    u.q[1] = *(const uint4*)p1;
    return u.v;
}

static __device__ __forceinline__ v8f zero8() {
    v8f z = {0.f,0.f,0.f,0.f,0.f,0.f,0.f,0.f};
    return z;
}

// Async copy of 16 bytes per lane: global -> LDS (tracked by ASYNCcnt).
// LDS aperture keeps the LDS byte offset in addr[31:0] of the generic pointer.
static __device__ __forceinline__ void async_ld16(const _Float16* gptr, _Float16* lptr) {
    unsigned lo = (unsigned)(uintptr_t)lptr;
    asm volatile("global_load_async_to_lds_b128 %0, %1, off"
                 :: "v"(lo), "v"(gptr) : "memory");
}

static __device__ __forceinline__ void wait_async0() {
    asm volatile("s_wait_asynccnt 0x0" ::: "memory");
}

// ---------------------------------------------------------------------------
// Weight convert + transpose: W[N][K] f32 -> Wt[K][N] f16
// ---------------------------------------------------------------------------
__global__ void transpose_f16(const float* __restrict__ W, _Float16* __restrict__ Wt,
                              int N, int K) {
    long long idx = (long long)blockIdx.x * 256 + threadIdx.x;
    if (idx >= (long long)N * K) return;
    int n = (int)(idx / K);
    int k = (int)(idx % K);
    Wt[(size_t)k * N + n] = (_Float16)W[(size_t)n * K + k];
}

// ---------------------------------------------------------------------------
// LayerNorm over D=768, output f16. One 256-thread block per row.
// ---------------------------------------------------------------------------
__global__ __launch_bounds__(256) void ln_kernel(const float* __restrict__ x,
                                                 const float* __restrict__ gam,
                                                 const float* __restrict__ bet,
                                                 _Float16* __restrict__ out) {
    __shared__ float s1[256], s2[256];
    const int row = blockIdx.x, tid = threadIdx.x;
    const float* xr = x + (size_t)row * Dd;
    float a = 0.f, b = 0.f;
    for (int i = tid; i < Dd; i += 256) { float v = xr[i]; a += v; b += v * v; }
    s1[tid] = a; s2[tid] = b;
    __syncthreads();
    for (int o = 128; o > 0; o >>= 1) {
        if (tid < o) { s1[tid] += s1[tid + o]; s2[tid] += s2[tid + o]; }
        __syncthreads();
    }
    float mean = s1[0] * (1.f / Dd);
    float var  = s2[0] * (1.f / Dd) - mean * mean;
    float rstd = rsqrtf(var + 1e-5f);
    for (int i = tid; i < Dd; i += 256) {
        float v = (xr[i] - mean) * rstd * gam[i] + bet[i];
        out[(size_t)row * Dd + i] = (_Float16)v;
    }
}

// ---------------------------------------------------------------------------
// WMMA GEMM: C[M,N] = A[M,K] @ Bt[K,N] (+bias) with mode-specific epilogue.
//   mode 0: QKV scatter to q16[B,H,T,HD] (scaled 1/8), kT16[B,H,HD,T], v16[B,H,T,HD]
//   mode 1: outf = acc + bias + resid   (attention out-proj + residual)
//   mode 2: outh = gelu(acc + bias)     (MLP fc1, f16 out)
//   mode 3: outf = acc + bias + resid   (MLP fc2 + residual -> d_out)
// Block tile 64x128, 8 waves (2x4), wave tile 32x32, K-stage 64,
// double-buffered LDS filled with global_load_async_to_lds_b128 (ASYNCcnt).
// ---------------------------------------------------------------------------
#define BM  64
#define BN  128
#define BKS 64

__global__ __launch_bounds__(256) void gemm_wmma(
    const _Float16* __restrict__ A, const _Float16* __restrict__ Bt,
    const float* __restrict__ bias, int M, int N, int K, int mode,
    const float* __restrict__ resid, float* __restrict__ outf,
    _Float16* __restrict__ outh,
    _Float16* __restrict__ q16, _Float16* __restrict__ kt16, _Float16* __restrict__ v16)
{
    __shared__ _Float16 sA[2][BM * BKS];   // 2 x 8 KB
    __shared__ _Float16 sB[2][BKS * BN];   // 2 x 16 KB

    const int tid  = threadIdx.x;
    const int lane = tid & 31;
    const int wave = tid >> 5;
    const int wm   = wave >> 2;      // 0..1
    const int wn   = wave & 3;       // 0..3
    const int g    = lane >> 4;
    const int nl   = lane & 15;
    const int bm   = blockIdx.y * BM;
    const int bn   = blockIdx.x * BN;

    v8f acc[2][2];
    acc[0][0] = zero8(); acc[0][1] = zero8(); acc[1][0] = zero8(); acc[1][1] = zero8();

    const int nst = K / BKS;

    // ---- stage 0 prologue ----
    {
#pragma unroll
        for (int i = 0; i < 2; ++i) {              // sA: 512 16B chunks
            int ci = tid + 256 * i;
            int r = ci >> 3, c = (ci & 7) * 8;
            async_ld16(&A[(size_t)(bm + r) * K + c], &sA[0][r * BKS + c]);
        }
#pragma unroll
        for (int i = 0; i < 4; ++i) {              // sB: 1024 16B chunks
            int ci = tid + 256 * i;
            int r = ci >> 4, c = (ci & 15) * 8;
            async_ld16(&Bt[(size_t)r * N + bn + c], &sB[0][r * BN + c]);
        }
    }
    wait_async0();
    __syncthreads();

    for (int s = 0; s < nst; ++s) {
        const int buf = s & 1;
        if (s + 1 < nst) {                          // prefetch next stage
            const int k0 = (s + 1) * BKS;
            const int nbuf = buf ^ 1;
#pragma unroll
            for (int i = 0; i < 2; ++i) {
                int ci = tid + 256 * i;
                int r = ci >> 3, c = (ci & 7) * 8;
                async_ld16(&A[(size_t)(bm + r) * K + k0 + c], &sA[nbuf][r * BKS + c]);
            }
#pragma unroll
            for (int i = 0; i < 4; ++i) {
                int ci = tid + 256 * i;
                int r = ci >> 4, c = (ci & 15) * 8;
                async_ld16(&Bt[(size_t)(k0 + r) * N + bn + c], &sB[nbuf][r * BN + c]);
            }
        }
        // ---- compute current stage: 2 K-substeps x 2x2 WMMA ----
        const _Float16* pA = sA[buf];
        const _Float16* pB = sB[buf];
#pragma unroll
        for (int ks = 0; ks < 2; ++ks) {
            v16h af[2], bf[2];
#pragma unroll
            for (int ms = 0; ms < 2; ++ms) {
                int r = wm * 32 + ms * 16 + nl;
                af[ms] = load_frag(&pA[r * BKS + ks * 32 + g * 8],
                                   &pA[r * BKS + ks * 32 + 16 + g * 8]);
            }
#pragma unroll
            for (int ns = 0; ns < 2; ++ns) {
                int c = wn * 32 + ns * 16;
#pragma unroll
                for (int e = 0; e < 16; ++e)
                    bf[ns][e] = pB[(ks * 32 + (e < 8 ? g * 8 + e : 8 + g * 8 + e)) * BN + c + nl];
            }
#pragma unroll
            for (int ms = 0; ms < 2; ++ms)
#pragma unroll
                for (int ns = 0; ns < 2; ++ns)
                    acc[ms][ns] = __builtin_amdgcn_wmma_f32_16x16x32_f16(
                        false, af[ms], false, bf[ns], (short)0, acc[ms][ns], false, false);
        }
        wait_async0();       // next-stage tiles resident
        __syncthreads();     // everyone done reading current buffers
    }

#pragma unroll
    for (int ms = 0; ms < 2; ++ms) {
#pragma unroll
        for (int ns = 0; ns < 2; ++ns) {
            const int gn = bn + wn * 32 + ns * 16 + nl;
            const float bb = bias ? bias[gn] : 0.f;
#pragma unroll
            for (int d = 0; d < 8; ++d) {
                const int gm = bm + wm * 32 + ms * 16 + d + 8 * g;
                float val = acc[ms][ns][d] + bb;
                if (mode == 0) {
                    int part = gn / Dd;           // 0=q, 1=k, 2=v
                    int c    = gn - part * Dd;
                    int h    = c >> 6;
                    int hd   = c & 63;
                    int b    = gm / Tt;
                    int t    = gm - b * Tt;
                    size_t bh = (size_t)(b * Hh + h);
                    if (part == 0)
                        q16[(bh * Tt + t) * HDim + hd] = (_Float16)(val * 0.125f);
                    else if (part == 1)
                        kt16[(bh * HDim + hd) * Tt + t] = (_Float16)val;
                    else
                        v16[(bh * Tt + t) * HDim + hd] = (_Float16)val;
                } else if (mode == 2) {
                    float ge = 0.5f * val * (1.f + erff(val * 0.70710678118f));
                    outh[(size_t)gm * N + gn] = (_Float16)ge;
                } else { // mode 1 / 3: residual add, f32 out
                    outf[(size_t)gm * Dd + gn] = val + resid[(size_t)gm * Dd + gn];
                }
            }
        }
    }
}

// ---------------------------------------------------------------------------
// Banded flash attention: one wave per (bh, 16-query tile).
// q pre-scaled by 1/sqrt(HD). Online softmax, key blocks of 32.
// ---------------------------------------------------------------------------
__global__ __launch_bounds__(32) void attn_kernel(
    const _Float16* __restrict__ q16, const _Float16* __restrict__ kt16,
    const _Float16* __restrict__ v16, _Float16* __restrict__ o16,
    const int* __restrict__ winp)
{
    __shared__ _Float16 sP[16 * 32];  // 1 KB: P tile C-layout -> A-layout staging
    const int lane = threadIdx.x;
    const int g  = lane >> 4;
    const int nl = lane & 15;
    const int q0 = blockIdx.x * 16;
    const int bh = blockIdx.y;              // b*H + h
    const int b  = bh / Hh;
    const int h  = bh - b * Hh;
    const int win = winp[0];

    const _Float16* qbase = q16 + ((size_t)bh * Tt + q0) * HDim;
    v16h qa[2];
#pragma unroll
    for (int s = 0; s < 2; ++s)
        qa[s] = load_frag(qbase + nl * HDim + s * 32 + g * 8,
                          qbase + nl * HDim + s * 32 + 16 + g * 8);

    v8f o[4];
    o[0] = zero8(); o[1] = zero8(); o[2] = zero8(); o[3] = zero8();
    float m_r[8], l_r[8];
#pragma unroll
    for (int d = 0; d < 8; ++d) { m_r[d] = -1e30f; l_r[d] = 0.f; }

    int klo = q0 - win;        if (klo < 0) klo = 0;
    int khi = q0 + 16 + win;   if (khi > Tt) khi = Tt;
    const _Float16* ktb = kt16 + (size_t)bh * HDim * Tt;

    for (int kb = (klo & ~31); kb < khi; kb += 32) {
        // ---- scores S = q @ K^T (two 16-key n-tiles) ----
        v8f sc[2];
#pragma unroll
        for (int ns = 0; ns < 2; ++ns) {
            v8f c = zero8();
#pragma unroll
            for (int s = 0; s < 2; ++s) {
                v16h bf;
#pragma unroll
                for (int e = 0; e < 16; ++e)
                    bf[e] = ktb[(size_t)(s * 32 + (e < 8 ? g * 8 + e : 8 + g * 8 + e)) * Tt + kb + ns * 16 + nl];
                c = __builtin_amdgcn_wmma_f32_16x16x32_f16(
                        false, qa[s], false, bf, (short)0, c, false, false);
            }
            sc[ns] = c;
        }
        // ---- mask + online softmax update ----
#pragma unroll
        for (int d = 0; d < 8; ++d) {
            const int qi  = q0 + d + 8 * g;
            const int k0i = kb + nl;
            const int k1i = kb + 16 + nl;
            const int d0  = (qi > k0i) ? (qi - k0i) : (k0i - qi);
            const int d1  = (qi > k1i) ? (qi - k1i) : (k1i - qi);
            const bool ok0 = (d0 <= win) && (k0i < Tt);
            const bool ok1 = (d1 <= win) && (k1i < Tt);
            float s0 = ok0 ? sc[0][d] : -1e30f;
            float s1 = ok1 ? sc[1][d] : -1e30f;
            float tmax = fmaxf(s0, s1);
#pragma unroll
            for (int off = 8; off > 0; off >>= 1)
                tmax = fmaxf(tmax, __shfl_xor(tmax, off));
            const float mnew  = fmaxf(m_r[d], tmax);
            const float alpha = __expf(m_r[d] - mnew);
            const float p0 = ok0 ? __expf(s0 - mnew) : 0.f;
            const float p1 = ok1 ? __expf(s1 - mnew) : 0.f;
            float ps = p0 + p1;
#pragma unroll
            for (int off = 8; off > 0; off >>= 1)
                ps += __shfl_xor(ps, off);
            l_r[d] = l_r[d] * alpha + ps;
            m_r[d] = mnew;
#pragma unroll
            for (int nv = 0; nv < 4; ++nv) o[nv][d] *= alpha;
            const int rr = d + 8 * g;
            sP[rr * 32 + nl]      = (_Float16)p0;
            sP[rr * 32 + 16 + nl] = (_Float16)p1;
        }
        __syncthreads();
        v16h pa = load_frag(&sP[nl * 32 + g * 8], &sP[nl * 32 + 16 + g * 8]);
        // ---- O += P @ V ----
#pragma unroll
        for (int nv = 0; nv < 4; ++nv) {
            Frag16 u;
#pragma unroll
            for (int e = 0; e < 16; ++e) {
                const int key = kb + (e < 8 ? g * 8 + e : 8 + g * 8 + e);
                u.v[e] = (key < Tt) ? v16[((size_t)bh * Tt + key) * HDim + nv * 16 + nl] : (_Float16)0.f;
            }
            o[nv] = __builtin_amdgcn_wmma_f32_16x16x32_f16(
                        false, pa, false, u.v, (short)0, o[nv], false, false);
        }
        __syncthreads();
    }
    // ---- normalize + write [token, D] f16 ----
#pragma unroll
    for (int d = 0; d < 8; ++d) {
        const float inv = 1.f / l_r[d];
        const int tok = b * Tt + q0 + d + 8 * g;
#pragma unroll
        for (int nv = 0; nv < 4; ++nv)
            o16[(size_t)tok * Dd + h * HDim + nv * 16 + nl] =
                (_Float16)(o[nv][d] * inv);
    }
}

// ---------------------------------------------------------------------------
extern "C" void kernel_launch(void* const* d_in, const int* in_sizes, int n_in,
                              void* d_out, int out_size, void* d_ws, size_t ws_size,
                              hipStream_t stream)
{
    const float* x    = (const float*)d_in[0];
    const float* Wqkv = (const float*)d_in[1];
    const float* bqkv = (const float*)d_in[2];
    const float* Wo   = (const float*)d_in[3];
    const float* bo   = (const float*)d_in[4];
    const float* W1   = (const float*)d_in[5];
    const float* bb1  = (const float*)d_in[6];
    const float* W2   = (const float*)d_in[7];
    const float* bb2  = (const float*)d_in[8];
    const float* g1   = (const float*)d_in[9];
    const float* be1  = (const float*)d_in[10];
    const float* g2   = (const float*)d_in[11];
    const float* be2  = (const float*)d_in[12];
    const int*   win  = (const int*)d_in[13];

    char* w = (char*)d_ws;
    auto carve = [&](size_t bytes) -> char* {
        char* p = w;
        w += (bytes + 255) & ~(size_t)255;
        return p;
    };
    _Float16* WqkvT = (_Float16*)carve((size_t)768 * 2304 * 2);
    _Float16* WoT   = (_Float16*)carve((size_t)768 * 768 * 2);
    _Float16* W1T   = (_Float16*)carve((size_t)768 * 3072 * 2);
    _Float16* W2T   = (_Float16*)carve((size_t)3072 * 768 * 2);
    _Float16* xn16  = (_Float16*)carve((size_t)NTOK * Dd * 2);
    _Float16* q16   = (_Float16*)carve((size_t)NTOK * Dd * 2);
    _Float16* kt16  = (_Float16*)carve((size_t)NTOK * Dd * 2);
    _Float16* v16   = (_Float16*)carve((size_t)NTOK * Dd * 2);
    _Float16* o16   = (_Float16*)carve((size_t)NTOK * Dd * 2);
    float*    x1    = (float*)   carve((size_t)NTOK * Dd * 4);
    _Float16* xn2   = (_Float16*)carve((size_t)NTOK * Dd * 2);
    _Float16* h16   = (_Float16*)carve((size_t)NTOK * MLPD * 2);

    auto cdiv = [](long long a, long long b) { return (int)((a + b - 1) / b); };

    // weight prep (f32 -> f16, [N,K] -> [K,N])
    transpose_f16<<<cdiv(2304LL * 768, 256), 256, 0, stream>>>(Wqkv, WqkvT, 2304, 768);
    transpose_f16<<<cdiv(768LL * 768, 256), 256, 0, stream>>>(Wo, WoT, 768, 768);
    transpose_f16<<<cdiv(3072LL * 768, 256), 256, 0, stream>>>(W1, W1T, 3072, 768);
    transpose_f16<<<cdiv(768LL * 3072, 256), 256, 0, stream>>>(W2, W2T, 768, 3072);

    // LN1 -> QKV -> attention
    ln_kernel<<<NTOK, 256, 0, stream>>>(x, g1, be1, xn16);
    gemm_wmma<<<dim3(2304 / BN, NTOK / BM), 256, 0, stream>>>(
        xn16, WqkvT, bqkv, NTOK, 2304, 768, 0,
        nullptr, nullptr, nullptr, q16, kt16, v16);
    attn_kernel<<<dim3(Tt / 16, Bsz * Hh), 32, 0, stream>>>(q16, kt16, v16, o16, win);

    // out-proj + residual -> x1
    gemm_wmma<<<dim3(768 / BN, NTOK / BM), 256, 0, stream>>>(
        o16, WoT, bo, NTOK, 768, 768, 1,
        x, x1, nullptr, nullptr, nullptr, nullptr);

    // LN2 -> MLP -> residual -> d_out
    ln_kernel<<<NTOK, 256, 0, stream>>>(x1, g2, be2, xn2);
    gemm_wmma<<<dim3(3072 / BN, NTOK / BM), 256, 0, stream>>>(
        xn2, W1T, bb1, NTOK, 3072, 768, 2,
        nullptr, nullptr, h16, nullptr, nullptr, nullptr);
    gemm_wmma<<<dim3(768 / BN, NTOK / BM), 256, 0, stream>>>(
        h16, W2T, bb2, NTOK, 768, 3072, 3,
        x1, (float*)d_out, nullptr, nullptr, nullptr, nullptr);
}
